// Quantize_42013370090101
// MI455X (gfx1250) — matrix operand order repaired and checked by reading, not compiled, not run on hardware
//
#include <hip/hip_runtime.h>

typedef __attribute__((ext_vector_type(16))) _Float16 v16h;
typedef __attribute__((ext_vector_type(8)))  float    v8f;

#define N_ROWS   262144        // 64 * 4096
#define DIM      64
#define KCB      512
#define CHUNK    256           // codebook entries staged in LDS at a time
#define CBS      72            // padded f16 stride per codebook row (144B, 16B-aligned pieces)
#define QELEMS   ((size_t)N_ROWS * DIM)   // 16777216

// workspace (floats): ws[0] = diff accumulator, ws[64..64+512) = ||w_k||^2
__global__ void vq_prep(const float* __restrict__ w, float* __restrict__ ws) {
    int t = blockIdx.x * blockDim.x + threadIdx.x;
    if (t == 0) ws[0] = 0.0f;
    if (t < KCB) {
        const float* wr = w + t * DIM;
        float s = 0.0f;
#pragma unroll
        for (int d = 0; d < DIM; ++d) { float v = wr[d]; s += v * v; }
        ws[64 + t] = s;
    }
}

// Each 256-thread block = 8 waves; each wave owns TWO 16-row M-tiles (32 rows).
// argmin(||x-w||^2) == argmax(x.w - ||w||^2/2): fold -||w||^2/2 into the WMMA C
// initializer so the post-WMMA chain is compare/select only. B fragments and the
// ||w||^2 scalar are software-pipelined one tile ahead to hide LDS latency and
// fill the WMMA->VALU hazard window.
__global__ __launch_bounds__(256)
void vq_main(const float* __restrict__ x, const float* __restrict__ w,
             const float* __restrict__ ws, float* __restrict__ out,
             float* __restrict__ acc) {
    __shared__ _Float16 cb[CHUNK * CBS];   // 36,864 B
    __shared__ float    swn[KCB];          //  2,048 B
    __shared__ int      sidx[256];         //  1,024 B

    const int tid  = threadIdx.x;
    const int wave = tid >> 5;
    const int lane = tid & 31;
    const int mn   = lane & 15;   // A: row-in-tile; B: col-in-tile; C: N index
    const int hi   = lane >> 4;   // half-wave select

    for (int i = tid; i < KCB; i += 256) swn[i] = ws[64 + i];

    // ---- A fragments: two 16-row tiles of x, f32 -> f16 ----
    const int rowBase = blockIdx.x * 256 + wave * 32;   // 32 rows per wave
    const float* xr0 = x + (size_t)(rowBase + mn) * DIM;
    const float* xr1 = x + (size_t)(rowBase + 16 + mn) * DIM;
    v16h a0, a1, a2, a3;
#pragma unroll
    for (int e = 0; e < 8; ++e) {
        a0[e]     = (_Float16)xr0[hi * 8 + e];        // K = hi*8 + e
        a0[8 + e] = (_Float16)xr0[16 + hi * 8 + e];   // K = 16 + hi*8 + e
        a1[e]     = (_Float16)xr0[32 + hi * 8 + e];
        a1[8 + e] = (_Float16)xr0[48 + hi * 8 + e];
        a2[e]     = (_Float16)xr1[hi * 8 + e];
        a2[8 + e] = (_Float16)xr1[16 + hi * 8 + e];
        a3[e]     = (_Float16)xr1[32 + hi * 8 + e];
        a3[8 + e] = (_Float16)xr1[48 + hi * 8 + e];
    }

    float best[16];
    int   bidx[16];
#pragma unroll
    for (int r = 0; r < 16; ++r) { best[r] = -3.0e38f; bidx[r] = 0; }

    // B 32x16 layout: lane = N (mn), halfs e<8 -> K = hi*8 + e, e>=8 -> K = 16 + hi*8 + (e-8) (k_base 0 / 32)
    const _Float16* cbb = &cb[mn * CBS + hi * 8];

    // ---- scan codebook in two LDS chunks of 256 entries ----
    for (int chunk = 0; chunk < KCB / CHUNK; ++chunk) {
        __syncthreads();   // previous chunk's compute done before overwrite
        const float* wsrc = w + (size_t)chunk * CHUNK * DIM;
        for (int i = tid; i < CHUNK * DIM; i += 256) {
            int n = i >> 6, d = i & 63;
            cb[n * CBS + d] = (_Float16)wsrc[i];
        }
        __syncthreads();

        // prologue: tile 0 fragments
        v16h b0, b1;
#pragma unroll
        for (int e = 0; e < 16; ++e) { b0[e] = cbb[e + (e & 8)]; b1[e] = cbb[32 + e + (e & 8)]; }
        float wn = swn[chunk * CHUNK + mn];

#pragma unroll 2
        for (int nt = 0; nt < CHUNK / 16; ++nt) {
            // prefetch next tile's B fragments + norm (independent of this tile)
            v16h nb0 = b0, nb1 = b1;
            float nwn = wn;
            if (nt < CHUNK / 16 - 1) {
                const _Float16* br = cbb + (nt + 1) * 16 * CBS;
#pragma unroll
                for (int e = 0; e < 16; ++e) { nb0[e] = br[e + (e & 8)]; nb1[e] = br[32 + e + (e & 8)]; }
                nwn = swn[chunk * CHUNK + (nt + 1) * 16 + mn];
            }

            const int   nidx = chunk * CHUNK + nt * 16 + mn;
            const float cini = -0.5f * wn;
            v8f c0, c1;
#pragma unroll
            for (int r = 0; r < 8; ++r) { c0[r] = cini; c1[r] = cini; }
            // two independent accumulator chains -> ILP across hazard window
            c0 = __builtin_amdgcn_wmma_f32_16x16x32_f16(false, a0, false, b0,
                                                        (short)0, c0, false, false);
            c1 = __builtin_amdgcn_wmma_f32_16x16x32_f16(false, a2, false, b0,
                                                        (short)0, c1, false, false);
            c0 = __builtin_amdgcn_wmma_f32_16x16x32_f16(false, a1, false, b1,
                                                        (short)0, c0, false, false);
            c1 = __builtin_amdgcn_wmma_f32_16x16x32_f16(false, a3, false, b1,
                                                        (short)0, c1, false, false);
#pragma unroll
            for (int r = 0; r < 8; ++r) {
                if (c0[r] > best[r])     { best[r]     = c0[r]; bidx[r]     = nidx; }
                if (c1[r] > best[8 + r]) { best[8 + r] = c1[r]; bidx[8 + r] = nidx; }
            }
            b0 = nb0; b1 = nb1; wn = nwn;
        }
    }

    // ---- argmax reduce across the 16 lanes of each half-wave ----
#pragma unroll
    for (int off = 1; off < 16; off <<= 1) {
#pragma unroll
        for (int r = 0; r < 16; ++r) {
            float ov = __shfl_xor(best[r], off, 32);
            int   oi = __shfl_xor(bidx[r], off, 32);
            if (ov > best[r] || (ov == best[r] && oi < bidx[r])) {
                best[r] = ov; bidx[r] = oi;
            }
        }
    }
    if (mn == 0) {
#pragma unroll
        for (int r = 0; r < 8; ++r) {            // C row m = hi*8 + r
            sidx[wave * 32 + hi * 8 + r]      = bidx[r];
            sidx[wave * 32 + 16 + hi * 8 + r] = bidx[8 + r];
        }
    }
    __syncthreads();

    // ---- gather winning codebook rows (f32), write quantize, diff, indices ----
    const int grow = rowBase + lane;                 // one full row per lane
    const int idx  = sidx[wave * 32 + lane];
    const float4* qsrc = (const float4*)(w + (size_t)idx * DIM);
    float4*       qdst = (float4*)(out + (size_t)grow * DIM);
    const float4* xsrc = (const float4*)(x + (size_t)grow * DIM);
    float lsum = 0.0f;
#pragma unroll
    for (int c4 = 0; c4 < 16; ++c4) {
        float4 wv = qsrc[c4];
        float4 xv = xsrc[c4];
        qdst[c4] = wv;
        float d0 = wv.x - xv.x, d1 = wv.y - xv.y, d2 = wv.z - xv.z, d3 = wv.w - xv.w;
        lsum += d0 * d0 + d1 * d1 + d2 * d2 + d3 * d3;
    }
    out[QELEMS + 1 + (size_t)grow] = (float)idx;     // embed_ind as float

#pragma unroll
    for (int off = 1; off < 32; off <<= 1) lsum += __shfl_xor(lsum, off, 32);
    if (lane == 0) atomicAdd(acc, lsum);
}

__global__ void vq_final(const float* __restrict__ ws, float* __restrict__ out) {
    out[QELEMS] = ws[0] / (float)QELEMS;
}

extern "C" void kernel_launch(void* const* d_in, const int* in_sizes, int n_in,
                              void* d_out, int out_size, void* d_ws, size_t ws_size,
                              hipStream_t stream) {
    const float* x = (const float*)d_in[0];        // [64,4096,64] f32
    const float* w = (const float*)d_in[1];        // [512,64] f32
    float* out = (float*)d_out;                    // quantize | diff | embed_ind
    float* ws  = (float*)d_ws;

    vq_prep<<<1, 512, 0, stream>>>(w, ws);
    vq_main<<<N_ROWS / 256, 256, 0, stream>>>(x, w, ws, out, ws);
    vq_final<<<1, 1, 0, stream>>>(ws, out);
}
